// VectorAndPatchDiscriminatorINv2_25563645346509
// MI455X (gfx1250) — matrix operand order repaired and checked by reading, not compiled, not run on hardware
//
#include <hip/hip_runtime.h>
#include <hip/hip_bf16.h>
#include <cmath>

typedef __bf16 bf16_t;
typedef __attribute__((ext_vector_type(16))) __bf16 v16bf;
typedef __attribute__((ext_vector_type(8)))  __bf16 v8bf;
typedef __attribute__((ext_vector_type(8)))  float  v8f;
typedef __attribute__((ext_vector_type(4)))  float  v4f;
typedef int v4i_t __attribute__((vector_size(16)));   // async-LDS builtin operand type

#define EPSN 1e-5f
#define INV_SQRT2F 0.70710678118654752f
#define MT 4        // 16x16 output tiles per wave along M (wave owns 64x16)
#define KSTEP 128   // staged K-window (bf16 elems); Cin % 128 == 0 on fast path

#define AS1 __attribute__((address_space(1)))
#define AS3 __attribute__((address_space(3)))

__device__ __forceinline__ float gelu_exact(float x) {
    return 0.5f * x * (1.0f + erff(x * INV_SQRT2F));
}

// ---------------------------------------------------------------------------
// Pack OIHW fp32 weights into [Cout][Kpad] bf16, K = kh*KW*Cin + kw*Cin + ci
// (ci fastest, matching NHWC activations), zero-padded to Kpad (mult of 32).
// ---------------------------------------------------------------------------
__global__ __launch_bounds__(256) void pack_w_kernel(
    const float* __restrict__ w, bf16_t* __restrict__ wp,
    int Cout, int Cin, int KH, int KW, int Kpad)
{
    int t = blockIdx.x * blockDim.x + threadIdx.x;
    int total = Cout * Kpad;
    if (t >= total) return;
    int co = t / Kpad;
    int k  = t - co * Kpad;
    int K  = Cin * KH * KW;
    float v = 0.0f;
    if (k < K) {
        int KWC = KW * Cin;
        int kh = k / KWC;
        int r  = k - kh * KWC;
        int kw = r / Cin;
        int ci = r - kw * Cin;
        v = w[(((size_t)co * Cin + ci) * KH + kh) * KW + kw];
    }
    wp[t] = (bf16_t)v;
}

// ---------------------------------------------------------------------------
// FAST implicit-GEMM conv (requires Cin % 128 == 0), bf16 in / fp32 out.
//   GEMM: M = Cout, N = B*OH*OW, K = Cin*KH*KW.
//   Block = 8 waves sharing a 64-row M tile; each wave owns a 64x16 output
//   tile (MT=4 accumulators) and its own 16-pixel N tile.
//   A tile (64 x 128 K, 16 KB) is cooperatively staged in LDS per K-window
//   via async global->LDS (ASYNCcnt / s_wait_asynccnt), then WMMAs are fed
//   with ds_load_b128. One B fragment feeds MT WMMAs.
// Requires: gridDim.x * 8 * 16 == N exactly (no early-exit: barriers).
// A-frag (16-bit 16x32): lane<16: M=lane, K=0..15; lane>=16: K=16..31.
// B-frag (16-bit 32x16): lane<16: N=lane, K=0..15; lane>=16: K=16..31.
// ---------------------------------------------------------------------------
__global__ __launch_bounds__(256) void conv_wmma_fast_kernel(
    const bf16_t* __restrict__ x,   // NHWC bf16 activations
    const bf16_t* __restrict__ wp,  // packed bf16 [Cout][K]
    float* __restrict__ y,          // NHWC fp32 raw conv output
    int Bn, int Cin, int H, int W, int Cout, int OH, int OW,
    int KH, int KW, int stride, int pad)
{
    __shared__ bf16_t lds_a[64 * KSTEP];          // 16 KB A tile

    const int tid  = threadIdx.x;
    const int lane = tid & 31;
    const int wave = tid >> 5;
    const int l16  = lane & 15;
    const int hi   = lane >> 4;                   // 0 or 1
    const int K    = Cin * KH * KW;
    const int m0   = blockIdx.y * (16 * MT);
    const int n0   = (blockIdx.x * 8 + wave) << 4;

    const int n  = n0 + l16;
    const int b  = n / (OH * OW);
    const int p  = n - b * (OH * OW);
    const int oh = p / OW;
    const int ow = p - oh * OW;
    const int ih0 = oh * stride - pad;
    const int iw0 = ow * stride - pad;

    // Cooperative stage assignment: thread copies 32 bf16 (64B) of one row.
    const int srow = tid >> 2;                    // 0..63
    const int squo = (tid & 3) * 32;              // 0,32,64,96
    const bf16_t* gsrc_row = wp + (size_t)(m0 + srow) * K + squo;
    bf16_t*       ldst     = lds_a + srow * KSTEP + squo;

    v8f acc[MT];
    #pragma unroll
    for (int t = 0; t < MT; ++t)
        #pragma unroll
        for (int i = 0; i < 8; ++i) acc[t][i] = 0.0f;

    const long long rowstride = (long long)W * Cin;
    const bf16_t* xbase = x + (long long)b * H * rowstride + hi * 16;

    for (int kh = 0; kh < KH; ++kh) {
        const int ih = ih0 + kh;
        const bool okh = (ih >= 0) && (ih < H);
        for (int kw = 0; kw < KW; ++kw) {
            const int iw = iw0 + kw;
            const bool ok = okh && (iw >= 0) && (iw < W);
            const bf16_t* xp = xbase + (long long)ih * rowstride + (long long)iw * Cin;
            const int kb = (kh * KW + kw) * Cin;

            for (int ci0 = 0; ci0 < Cin; ci0 += KSTEP) {
                // ---- stage A[64][128] into LDS (async DMA path) ----
                const bf16_t* gsrc = gsrc_row + kb + ci0;
#if __has_builtin(__builtin_amdgcn_global_load_async_to_lds_b128)
                {
                    AS1 v4i_t* gs = (AS1 v4i_t*)gsrc;
                    AS3 v4i_t* ld = (AS3 v4i_t*)ldst;
                    __builtin_amdgcn_global_load_async_to_lds_b128(gs, ld, 0,  0);
                    __builtin_amdgcn_global_load_async_to_lds_b128(gs, ld, 16, 0);
                    __builtin_amdgcn_global_load_async_to_lds_b128(gs, ld, 32, 0);
                    __builtin_amdgcn_global_load_async_to_lds_b128(gs, ld, 48, 0);
                }
#if __has_builtin(__builtin_amdgcn_s_wait_asynccnt)
                __builtin_amdgcn_s_wait_asynccnt(0);
#else
                asm volatile("s_wait_asynccnt 0x0" ::: "memory");
#endif
#else
                {   // synchronous fallback: 4x16B through VGPRs
                    v8bf t0 = *(const v8bf*)(gsrc);
                    v8bf t1 = *(const v8bf*)(gsrc + 8);
                    v8bf t2 = *(const v8bf*)(gsrc + 16);
                    v8bf t3 = *(const v8bf*)(gsrc + 24);
                    *(v8bf*)(ldst)      = t0;
                    *(v8bf*)(ldst + 8)  = t1;
                    *(v8bf*)(ldst + 16) = t2;
                    *(v8bf*)(ldst + 24) = t3;
                }
#endif
                __syncthreads();

                // ---- 4 WMMA K-chunks of 32 over the staged window ----
                #pragma unroll
                for (int kk = 0; kk < 4; ++kk) {
                    // B fragment: 16 consecutive bf16 channels, 2x16B loads
                    v16bf Bv;
                    if (ok) {
                        const bf16_t* bp = xp + ci0 + kk * 32;
                        v8bf b0 = *(const v8bf*)(bp);
                        v8bf b1 = *(const v8bf*)(bp + 8);
                        #pragma unroll
                        for (int i = 0; i < 8; ++i) { Bv[i] = b0[i]; Bv[i + 8] = b1[i]; }
                    } else {
                        #pragma unroll
                        for (int i = 0; i < 16; ++i) Bv[i] = (bf16_t)0.0f;
                    }
                    // MT A fragments from LDS reuse this B fragment
                    #pragma unroll
                    for (int t = 0; t < MT; ++t) {
                        const bf16_t* aq = lds_a + (t * 16 + l16) * KSTEP
                                         + kk * 32 + hi * 16;
                        v8bf a0 = *(const v8bf*)(aq);
                        v8bf a1 = *(const v8bf*)(aq + 8);
                        v16bf Av;
                        #pragma unroll
                        for (int i = 0; i < 8; ++i) { Av[i] = a0[i]; Av[i + 8] = a1[i]; }
                        acc[t] = __builtin_amdgcn_wmma_f32_16x16x32_bf16(
                            false, Av, false, Bv, (short)0, acc[t], false, false);
                    }
                }
                __syncthreads();   // protect LDS tile before next stage
            }
        }
    }

    // D-frag store: lane owns 8 consecutive channels at pixel n (NHWC).
    #pragma unroll
    for (int t = 0; t < MT; ++t) {
        float* out = y + (size_t)n * Cout + m0 + t * 16 + hi * 8;
        v4f s0 = {acc[t][0], acc[t][1], acc[t][2], acc[t][3]};
        v4f s1 = {acc[t][4], acc[t][5], acc[t][6], acc[t][7]};
        *(v4f*)out       = s0;
        *(v4f*)(out + 4) = s1;
    }
}

// ---------------------------------------------------------------------------
// GENERIC implicit-GEMM conv (any Cin; used only for Cin==1 layers: conv1a,
// res1 -- negligible FLOP share). fp32 input (the network image), fp32 out.
// ---------------------------------------------------------------------------
__global__ __launch_bounds__(256) void conv_wmma_generic_kernel(
    const float* __restrict__ x, const bf16_t* __restrict__ wp,
    float* __restrict__ y,
    int Bn, int Cin, int H, int W, int Cout, int OH, int OW,
    int KH, int KW, int stride, int pad, int Kpad)
{
    const int lane = threadIdx.x & 31;
    const int wave = threadIdx.x >> 5;
    const int l16  = lane & 15;
    const int hi   = lane >> 4;
    const int N    = Bn * OH * OW;
    const int m0   = blockIdx.y << 4;
    const int n0   = (blockIdx.x * 8 + wave) << 4;
    if (n0 >= N) return;

    const int K   = Cin * KH * KW;
    const int KWC = KW * Cin;
    const bf16_t* wrow = wp + (size_t)(m0 + l16) * Kpad;

    const int n   = n0 + l16;
    const int b   = n / (OH * OW);
    const int p   = n - b * (OH * OW);
    const int oh  = p / OW;
    const int ow  = p - oh * OW;
    const int ih0 = oh * stride - pad;
    const int iw0 = ow * stride - pad;

    v8f acc;
    #pragma unroll
    for (int i = 0; i < 8; ++i) acc[i] = 0.0f;

    for (int k0 = 0; k0 < Kpad; k0 += 32) {
        const int ka = k0 + hi * 16;
        v8bf alo = *(const v8bf*)(wrow + ka);
        v8bf ahi = *(const v8bf*)(wrow + ka + 8);
        v16bf A;
        #pragma unroll
        for (int i = 0; i < 8; ++i) { A[i] = alo[i]; A[i + 8] = ahi[i]; }

        const int kb = k0 + hi * 16;
        v16bf Bv;
        #pragma unroll
        for (int i = 0; i < 16; ++i) {
            const int k = kb + i;
            float v = 0.0f;
            if (k < K) {
                const int kh = k / KWC;
                const int r  = k - kh * KWC;
                const int kw = r / Cin;
                const int ci = r - kw * Cin;
                const int ih = ih0 + kh, iw = iw0 + kw;
                if (ih >= 0 && ih < H && iw >= 0 && iw < W)
                    v = x[((size_t)(b * H + ih) * W + iw) * Cin + ci];
            }
            Bv[i] = (bf16_t)v;
        }

        acc = __builtin_amdgcn_wmma_f32_16x16x32_bf16(
            false, A, false, Bv, (short)0, acc, false, false);
    }

    float* out = y + (size_t)n * Cout + m0 + hi * 8;
    v4f s0 = {acc[0], acc[1], acc[2], acc[3]};
    v4f s1 = {acc[4], acc[5], acc[6], acc[7]};
    *(v4f*)out       = s0;
    *(v4f*)(out + 4) = s1;
}

// ---------------------------------------------------------------------------
// Instance-norm stats over NHWC: one block handles (b, 64 channels).
// Coalesced: consecutive lanes read consecutive channels. LDS reduce.
// ---------------------------------------------------------------------------
__global__ __launch_bounds__(256) void stats_nhwc_kernel(
    const float* __restrict__ x, float* __restrict__ mean, float* __restrict__ rstd,
    int C, int HW)
{
    __shared__ float s_sum[256];
    __shared__ float s_sq[256];
    const int cblks = C >> 6;
    const int b  = blockIdx.x / cblks;
    const int c0 = (blockIdx.x - b * cblks) << 6;
    const int cl = threadIdx.x & 63;
    const int pr = threadIdx.x >> 6;
    const int c  = c0 + cl;
    const float* base = x + (size_t)b * HW * C + c;
    float sum = 0.f, sq = 0.f;
    for (int p = pr; p < HW; p += 4) {
        float v = base[(size_t)p * C];
        sum += v; sq += v * v;
    }
    s_sum[threadIdx.x] = sum;
    s_sq[threadIdx.x]  = sq;
    __syncthreads();
    if (pr == 0) {
        float ts = s_sum[cl] + s_sum[cl + 64] + s_sum[cl + 128] + s_sum[cl + 192];
        float tq = s_sq[cl]  + s_sq[cl + 64]  + s_sq[cl + 128]  + s_sq[cl + 192];
        float mn  = ts / (float)HW;
        float var = tq / (float)HW - mn * mn;
        mean[b * C + c] = mn;
        rstd[b * C + c] = rsqrtf(var + EPSN);
    }
}

// ---------------------------------------------------------------------------
// h_bf16 = gelu((x - mean) * rstd)   fp32 raw -> bf16 activation (conv feed)
// ---------------------------------------------------------------------------
__global__ __launch_bounds__(256) void norm_gelu_kernel(
    const float* __restrict__ x, bf16_t* __restrict__ xb,
    const float* __restrict__ mean, const float* __restrict__ rstd,
    int total, int C, int HW)
{
    int i = blockIdx.x * blockDim.x + threadIdx.x;
    if (i >= total) return;
    int c = i % C;
    int b = i / (HW * C);
    float v = (x[i] - mean[b * C + c]) * rstd[b * C + c];
    xb[i] = (bf16_t)gelu_exact(v);
}

// ---------------------------------------------------------------------------
// Residual-path inorm stats at FULL input resolution, recomputing the 1x1
// conv on the fly (value produced separately by the WMMA GEMM at stride s:
// conv1x1 stride-2 == subsample of stride-1). Res weight row staged in LDS.
// ---------------------------------------------------------------------------
__global__ __launch_bounds__(256) void res_stats_kernel(
    const float* __restrict__ x, const float* __restrict__ wres,
    float* __restrict__ mean, float* __restrict__ rstd,
    int Cin, int Cout, int HW)
{
    __shared__ float wsh[1024];
    __shared__ float s_sum[256];
    __shared__ float s_sq[256];
    const int b  = blockIdx.x / Cout;
    const int co = blockIdx.x - b * Cout;
    for (int ci = threadIdx.x; ci < Cin; ci += 256)
        wsh[ci] = wres[(size_t)co * Cin + ci];
    __syncthreads();
    float sum = 0.f, sq = 0.f;
    const float* xb = x + (size_t)b * HW * Cin;
    for (int p = threadIdx.x; p < HW; p += 256) {
        const float* xp = xb + (size_t)p * Cin;
        float d = 0.f;
        for (int ci = 0; ci < Cin; ++ci) d += wsh[ci] * xp[ci];
        sum += d; sq += d * d;
    }
    s_sum[threadIdx.x] = sum;
    s_sq[threadIdx.x]  = sq;
    __syncthreads();
    for (int s = 128; s > 0; s >>= 1) {
        if (threadIdx.x < s) {
            s_sum[threadIdx.x] += s_sum[threadIdx.x + s];
            s_sq[threadIdx.x]  += s_sq[threadIdx.x + s];
        }
        __syncthreads();
    }
    if (threadIdx.x == 0) {
        float mn  = s_sum[0] / (float)HW;
        float var = s_sq[0] / (float)HW - mn * mn;
        mean[blockIdx.x] = mn;
        rstd[blockIdx.x] = rsqrtf(var + EPSN);
    }
}

// ---------------------------------------------------------------------------
// x_next = gelu((norm(h2) + norm(res)) / sqrt(2)); writes fp32 AND bf16.
// ---------------------------------------------------------------------------
__global__ __launch_bounds__(256) void tail_kernel(
    const float* __restrict__ h2, const float* __restrict__ mh, const float* __restrict__ rh,
    const float* __restrict__ rs, const float* __restrict__ mr, const float* __restrict__ rr,
    float* __restrict__ out, bf16_t* __restrict__ outb,
    int total, int C, int HW)
{
    int i = blockIdx.x * blockDim.x + threadIdx.x;
    if (i >= total) return;
    int c = i % C;
    int b = i / (HW * C);
    int sc = b * C + c;
    float hv = (h2[i] - mh[sc]) * rh[sc];
    float rv = (rs[i] - mr[sc]) * rr[sc];
    float o  = gelu_exact((hv + rv) * INV_SQRT2F);
    out[i]  = o;
    outb[i] = (bf16_t)o;
}

// ---------------------------------------------------------------------------
// vec[b][n] = sum_f x5[b][f][n] * fcw[n][f] + fcb[n]   (x5 NHWC, f=128 pix)
// ---------------------------------------------------------------------------
__global__ __launch_bounds__(256) void fc_kernel(
    const float* __restrict__ x5, const float* __restrict__ fcw,
    const float* __restrict__ fcb, float* __restrict__ out)
{
    int b = blockIdx.x;
    const float* xr = x5 + (size_t)b * 128 * 2048;
    for (int n = threadIdx.x; n < 2048; n += 256) {
        float acc = fcb[n];
        const float* wr = fcw + (size_t)n * 128;
        for (int f = 0; f < 128; ++f)
            acc += xr[(size_t)f * 2048 + n] * wr[f];
        out[(size_t)b * 2048 + n] = acc;
    }
}

// ---------------------------------------------------------------------------
// patch[b,0,oh,ow] = sum_{ci,kh,kw} feat[b,oh-2+kh,ow-2+kw,ci]*wrf[ci,kh,kw]
// feat NHWC (16,4,32,1024); pad ((2,1),(2,1)); one block per output pixel.
// ---------------------------------------------------------------------------
__global__ __launch_bounds__(256) void patch_kernel(
    const float* __restrict__ feat, const float* __restrict__ wrf,
    float* __restrict__ out)
{
    __shared__ float red[256];
    int o  = blockIdx.x;          // b*128 + oh*32 + ow
    int b  = o >> 7;
    int p  = o & 127;
    int oh = p >> 5;
    int ow = p & 31;
    float acc = 0.f;
    for (int t = threadIdx.x; t < 1024 * 16; t += 256) {
        int ci = t >> 4;
        int kk = t & 15;
        int kh = kk >> 2;
        int kw = kk & 3;
        int ih = oh - 2 + kh, iw = ow - 2 + kw;
        if (ih >= 0 && ih < 4 && iw >= 0 && iw < 32)
            acc += feat[((size_t)(b * 4 + ih) * 32 + iw) * 1024 + ci]
                 * wrf[ci * 16 + kk];
    }
    red[threadIdx.x] = acc;
    __syncthreads();
    for (int s = 128; s > 0; s >>= 1) {
        if (threadIdx.x < s) red[threadIdx.x] += red[threadIdx.x + s];
        __syncthreads();
    }
    if (threadIdx.x == 0) out[o] = red[0];
}

// ---------------------------------------------------------------------------
// Host orchestration
// ---------------------------------------------------------------------------
static inline void run_conv_fast(hipStream_t s, const bf16_t* x, const float* w,
                                 bf16_t* wpk, float* y,
                                 int B, int Cin, int H, int W, int Cout,
                                 int OH, int OW, int KH, int KW, int stride, int pad)
{
    int K = Cin * KH * KW;      // Cin % 128 == 0 -> K % 128 == 0
    int tw = Cout * K;
    pack_w_kernel<<<(tw + 255) / 256, 256, 0, s>>>(w, wpk, Cout, Cin, KH, KW, K);
    int N = B * OH * OW;        // N/16 divisible by 8 for all layers here
    dim3 grid((N / 16) / 8, Cout / (16 * MT));
    conv_wmma_fast_kernel<<<grid, 256, 0, s>>>(x, wpk, y, B, Cin, H, W, Cout,
                                               OH, OW, KH, KW, stride, pad);
}

static inline void run_conv_generic(hipStream_t s, const float* x, const float* w,
                                    bf16_t* wpk, float* y,
                                    int B, int Cin, int H, int W, int Cout,
                                    int OH, int OW, int KH, int KW, int stride, int pad)
{
    int K    = Cin * KH * KW;
    int Kpad = (K + 31) & ~31;
    int tw   = Cout * Kpad;
    pack_w_kernel<<<(tw + 255) / 256, 256, 0, s>>>(w, wpk, Cout, Cin, KH, KW, Kpad);
    int N = B * OH * OW;
    dim3 grid((N / 16 + 7) / 8, Cout / 16);
    conv_wmma_generic_kernel<<<grid, 256, 0, s>>>(x, wpk, y, B, Cin, H, W, Cout,
                                                  OH, OW, KH, KW, stride, pad, Kpad);
}

extern "C" void kernel_launch(void* const* d_in, const int* in_sizes, int n_in,
                              void* d_out, int out_size, void* d_ws, size_t ws_size,
                              hipStream_t stream)
{
    (void)in_sizes; (void)n_in; (void)out_size; (void)ws_size;
    const int B = 16;

    const float* img = (const float*)d_in[0];
    const float* fcw = (const float*)d_in[16];
    const float* fcb = (const float*)d_in[17];
    const float* wrf = (const float*)d_in[18];

    // Workspace layout (256B aligned slices)
    char* ws = (char*)d_ws;
    size_t off = 0;
    auto take = [&](size_t bytes) -> char* {
        char* p = ws + off;
        off = (off + bytes + 255) & ~(size_t)255;
        return p;
    };
    const size_t ACT_ELEMS = (size_t)16 * 128 * 32 * 256; // 16.78M (max act)
    float*  xbuf0 = (float*)take(ACT_ELEMS * 4);   // fp32 x ping
    float*  xbuf1 = (float*)take(ACT_ELEMS * 4);   // fp32 x pong
    bf16_t* xb0   = (bf16_t*)take(ACT_ELEMS * 2);  // bf16 x ping
    bf16_t* xb1   = (bf16_t*)take(ACT_ELEMS * 2);  // bf16 x pong
    float*  t1    = (float*)take(ACT_ELEMS * 4);   // convA raw
    bf16_t* h1b   = (bf16_t*)take(ACT_ELEMS * 2);  // gelu(norm(convA)) bf16
    float*  t2    = (float*)take(ACT_ELEMS * 4);   // convB raw
    float*  t3    = (float*)take(ACT_ELEMS * 4);   // res (subsampled) raw
    bf16_t* wpk   = (bf16_t*)take((size_t)2048 * 18432 * 2); // largest packed W
    const size_t STAT_ELEMS = (size_t)16 * 2048;
    float* mh1 = (float*)take(STAT_ELEMS * 4);
    float* rh1 = (float*)take(STAT_ELEMS * 4);
    float* mh2 = (float*)take(STAT_ELEMS * 4);
    float* rh2 = (float*)take(STAT_ELEMS * 4);
    float* mr  = (float*)take(STAT_ELEMS * 4);
    float* rr  = (float*)take(STAT_ELEMS * 4);

    // Per-block configuration
    const int CinA[5]  = {1, 128, 256, 512, 1024};
    const int CoutA[5] = {128, 256, 512, 1024, 2048};
    const int Hin[5]   = {64, 32, 16, 8, 4};
    const int Win[5]   = {512, 256, 128, 64, 32};
    const int Str[5]   = {2, 2, 2, 2, 1};
    const int KA[5]    = {4, 4, 4, 4, 3};

    const float*  xcur  = img;     // fp32 view (NCHW C=1 == NHWC)
    const bf16_t* xcurb = nullptr; // bf16 view (blocks >= 2)
    float*  pong [2] = {xbuf0, xbuf1};
    bf16_t* pongb[2] = {xb0, xb1};
    const float* feat = nullptr;

    for (int i = 0; i < 5; ++i) {
        const float* wres = (const float*)d_in[1 + 3 * i];
        const float* wa   = (const float*)d_in[2 + 3 * i];
        const float* wb   = (const float*)d_in[3 + 3 * i];
        const int Cin = CinA[i], Cout = CoutA[i];
        const int H = Hin[i], W = Win[i], s = Str[i];
        const int OH = H / s, OW = W / s;
        const int HWf = H * W, HWo = OH * OW;
        const int total = B * HWo * Cout;
        float*  xnext  = pong [i & 1];
        bf16_t* xnextb = pongb[i & 1];

        // h = gelu(inorm(convA(x)))  -> bf16
        if (i == 0)
            run_conv_generic(stream, xcur, wa, wpk, t1, B, Cin, H, W, Cout,
                             OH, OW, KA[i], KA[i], s, 1);
        else
            run_conv_fast(stream, xcurb, wa, wpk, t1, B, Cin, H, W, Cout,
                          OH, OW, KA[i], KA[i], s, 1);
        stats_nhwc_kernel<<<B * (Cout / 64), 256, 0, stream>>>(t1, mh1, rh1, Cout, HWo);
        norm_gelu_kernel<<<(total + 255) / 256, 256, 0, stream>>>(
            t1, h1b, mh1, rh1, total, Cout, HWo);

        // h2 = convB(h) (3x3 SAME); stats only, norm fused into tail
        run_conv_fast(stream, h1b, wb, wpk, t2, B, Cout, OH, OW, Cout,
                      OH, OW, 3, 3, 1, 1);
        stats_nhwc_kernel<<<B * (Cout / 64), 256, 0, stream>>>(t2, mh2, rh2, Cout, HWo);

        // res: 1x1 conv at stride s == subsampled stride-1 conv (values);
        // stats recomputed at full input resolution from fp32 x.
        if (i == 0)
            run_conv_generic(stream, xcur, wres, wpk, t3, B, Cin, H, W, Cout,
                             OH, OW, 1, 1, s, 0);
        else
            run_conv_fast(stream, xcurb, wres, wpk, t3, B, Cin, H, W, Cout,
                          OH, OW, 1, 1, s, 0);
        res_stats_kernel<<<B * Cout, 256, 0, stream>>>(xcur, wres, mr, rr,
                                                       Cin, Cout, HWf);

        // x_next = gelu((norm(h2) + norm(res)) / sqrt(2))  -> fp32 + bf16
        tail_kernel<<<(total + 255) / 256, 256, 0, stream>>>(
            t2, mh2, rh2, t3, mr, rr, xnext, xnextb, total, Cout, HWo);

        if (i == 3) feat = xnext;         // (16,4,32,1024) NHWC fp32
        xcur  = xnext;
        xcurb = xnextb;
    }

    float* vec_out   = (float*)d_out;                 // 16*2048
    float* patch_out = (float*)d_out + 16 * 2048;     // 16*1*4*32

    fc_kernel<<<B, 256, 0, stream>>>(xcur, fcw, fcb, vec_out);
    patch_kernel<<<B * 128, 256, 0, stream>>>(feat, wrf, patch_out);
}
